// BasicBlock_38637525794932
// MI455X (gfx1250) — 8-bit matrix operand order repaired and checked by reading, compiled, not run on hardware
//
#include <hip/hip_runtime.h>
#include <hip/hip_bf16.h>
#include <cstdint>
#include <cstddef>


typedef __attribute__((ext_vector_type(8))) int v8i;

#define N_IMG 32
#define CCH   256
#define HIMG  28
#define WIMG  28
#define HWP   (HIMG * WIMG)
#define MTOT  (N_IMG * HWP)
#define HPAD  (HIMG + 2)
#define PADVOL ((size_t)N_IMG * HPAD * HPAD * CCH)
#define WVOL   ((size_t)CCH * CCH * 9)
#define BN_EPS 1e-5f

__device__ __forceinline__ int8_t sgn8(float f) {
    return (int8_t)((f > 0.0f) - (f < 0.0f));
}

__global__ __launch_bounds__(256) void binarize_x_kernel(
    const float* __restrict__ x, int8_t* __restrict__ xb) {
    int idx = blockIdx.x * 256 + threadIdx.x;
    int c  = idx & 255;
    int t  = idx >> 8;
    int wp = t % HPAD; t /= HPAD;
    int hp = t % HPAD;
    int n  = t / HPAD;
    int8_t v = 0;
    if (hp >= 1 && hp <= HIMG && wp >= 1 && wp <= WIMG) {
        float f = x[(((size_t)n * CCH + c) * HIMG + (hp - 1)) * WIMG + (wp - 1)];
        v = sgn8(f);
    }
    xb[idx] = v;
}

__global__ __launch_bounds__(256) void binarize_w_kernel(
    const float* __restrict__ w, int8_t* __restrict__ wbt) {
    int idx = blockIdx.x * 256 + threadIdx.x;
    int kw = idx % 3; int t = idx / 3;
    int kh = t % 3;   t /= 3;
    int c  = t % CCH; int o = t / CCH;
    int tap = kh * 3 + kw;
    wbt[(size_t)tap * (CCH * CCH) + (size_t)o * CCH + c] = sgn8(w[idx]);
}

__global__ __launch_bounds__(256) void binconv_wmma_kernel(
    const int8_t* __restrict__ xb,
    const int8_t* __restrict__ wbt,
    const float*  __restrict__ residual,
    float* __restrict__ outf)
{
    const int lane = threadIdx.x & 31;
    const int wave = threadIdx.x >> 5;
    const int r    = lane & 15;
    const int half = lane >> 4;

    const int m_strip = blockIdx.x * 128 + wave * 16;
    const int o_base  = blockIdx.y * 64;

    const int m   = m_strip + r;
    const int n_i = m / HWP;
    const int rem = m - n_i * HWP;
    const int h   = rem / WIMG;
    const int w   = rem - h * WIMG;

    v8i acc[4] = {};

    #pragma unroll
    for (int kh = 0; kh < 3; ++kh) {
        #pragma unroll
        for (int kw = 0; kw < 3; ++kw) {
            const int tap = kh * 3 + kw;
            const int8_t* arow =
                xb + ((size_t)((n_i * HPAD) + (h + kh)) * HPAD + (w + kw)) * CCH;
            const int8_t* wtap = wbt + (size_t)tap * (CCH * CCH);
            #pragma unroll
            for (int j = 0; j < 4; ++j) {
                v8i a;
                {
                    const int8_t* ap = arow + j * 64 + half * 16;
                    int2 p0 = *(const int2*)(ap +  0);
                    int2 p1 = *(const int2*)(ap +  8);
                    int2 p2 = *(const int2*)(ap + 32);
                    int2 p3 = *(const int2*)(ap + 40);
                    a[0] = p0.x; a[1] = p0.y; a[2] = p1.x; a[3] = p1.y;
                    a[4] = p2.x; a[5] = p2.y; a[6] = p3.x; a[7] = p3.y;
                }
                #pragma unroll
                for (int nt = 0; nt < 4; ++nt) {
                    const int8_t* bp =
                        wtap + (size_t)(o_base + nt * 16 + r) * CCH + j * 64 + half * 16;
                    int4 q0 = *(const int4*)(bp +  0);
                    int4 q1 = *(const int4*)(bp + 32);
                    v8i b;
                    b[0] = q0.x; b[1] = q0.y; b[2] = q0.z; b[3] = q0.w;
                    b[4] = q1.x; b[5] = q1.y; b[6] = q1.z; b[7] = q1.w;
                    acc[nt] = __builtin_amdgcn_wmma_i32_16x16x64_iu8(
                        true, a, true, b, acc[nt], false, false);
                }
            }
        }
    }

    #pragma unroll
    for (int nt = 0; nt < 4; ++nt) {
        const int o = o_base + nt * 16 + r;
        #pragma unroll
        for (int i = 0; i < 8; ++i) {
            const int mo = m_strip + half * 8 + i;
            float v = (float)acc[nt][i];
            if (residual != nullptr) {
                const int ni2 = mo / HWP;
                const int rr  = mo - ni2 * HWP;
                const int h2  = rr / WIMG;
                const int w2  = rr - h2 * WIMG;
                v += residual[(((size_t)ni2 * CCH + o) * HIMG + h2) * WIMG + w2];
            }
            outf[(size_t)mo * CCH + o] = v;
        }
    }
}

__global__ __launch_bounds__(256) void channel_stats_kernel(
    const float* __restrict__ outf, float* __restrict__ stats) {
    __shared__ float s1[256];
    __shared__ float s2[256];
    const int c = blockIdx.x;
    const int t = threadIdx.x;
    float a = 0.0f, b = 0.0f;
    for (int m = t; m < MTOT; m += 256) {
        float v = outf[(size_t)m * CCH + c];
        a += v;
        b += v * v;
    }
    s1[t] = a; s2[t] = b;
    __syncthreads();
    for (int s = 128; s > 0; s >>= 1) {
        if (t < s) { s1[t] += s1[t + s]; s2[t] += s2[t + s]; }
        __syncthreads();
    }
    if (t == 0) { stats[c] = s1[0]; stats[CCH + c] = s2[0]; }
}

__global__ __launch_bounds__(256) void bn_binarize_kernel(
    const float* __restrict__ outf, const float* __restrict__ stats,
    const float* __restrict__ gamma, const float* __restrict__ beta,
    int8_t* __restrict__ xb) {
    int idx = blockIdx.x * 256 + threadIdx.x;
    int c  = idx & 255;
    int t  = idx >> 8;
    int wp = t % HPAD; t /= HPAD;
    int hp = t % HPAD;
    int n  = t / HPAD;
    int8_t v = 0;
    if (hp >= 1 && hp <= HIMG && wp >= 1 && wp <= WIMG) {
        const int m = n * HWP + (hp - 1) * WIMG + (wp - 1);
        const float invM = 1.0f / (float)MTOT;
        float mean = stats[c] * invM;
        float var  = stats[CCH + c] * invM - mean * mean;
        float y = (outf[(size_t)m * CCH + c] - mean) * rsqrtf(var + BN_EPS)
                    * gamma[c] + beta[c];
        v = sgn8(y);
    }
    xb[idx] = v;
}

__global__ __launch_bounds__(256) void bn_out_kernel(
    const float* __restrict__ outf, const float* __restrict__ stats,
    const float* __restrict__ gamma, const float* __restrict__ beta,
    float* __restrict__ out) {
    int idx = blockIdx.x * 256 + threadIdx.x;
    int w = idx % WIMG; int t = idx / WIMG;
    int h = t % HIMG;   t /= HIMG;
    int c = t % CCH;    int n = t / CCH;
    const int m = n * HWP + h * WIMG + w;
    const float invM = 1.0f / (float)MTOT;
    float mean = stats[c] * invM;
    float var  = stats[CCH + c] * invM - mean * mean;
    float y = (outf[(size_t)m * CCH + c] - mean) * rsqrtf(var + BN_EPS)
                * gamma[c] + beta[c];
    out[idx] = fminf(1.0f, fmaxf(-1.0f, y));
}

extern "C" void kernel_launch(void* const* d_in, const int* in_sizes, int n_in,
                              void* d_out, int out_size, void* d_ws, size_t ws_size,
                              hipStream_t stream) {
    (void)in_sizes; (void)n_in; (void)out_size; (void)ws_size;
    const float* x  = (const float*)d_in[0];
    const float* w1 = (const float*)d_in[1];
    const float* g1 = (const float*)d_in[2];
    const float* b1 = (const float*)d_in[3];
    const float* w2 = (const float*)d_in[4];
    const float* g2 = (const float*)d_in[5];
    const float* b2 = (const float*)d_in[6];
    float* out = (float*)d_out;

    uint8_t* ws   = (uint8_t*)d_ws;
    int8_t*  xb   = (int8_t*)ws;
    int8_t*  wbt1 = (int8_t*)(ws + PADVOL);
    int8_t*  wbt2 = wbt1 + WVOL;
    float*   outf = (float*)(ws + PADVOL + 2 * WVOL);
    float*   stats = outf + (size_t)MTOT * CCH;

    const int padBlocks = (int)(PADVOL / 256);
    const int wBlocks   = (int)(WVOL / 256);
    const dim3 gemmGrid(MTOT / 128, CCH / 64);

    binarize_x_kernel<<<padBlocks, 256, 0, stream>>>(x, xb);
    binarize_w_kernel<<<wBlocks, 256, 0, stream>>>(w1, wbt1);
    binarize_w_kernel<<<wBlocks, 256, 0, stream>>>(w2, wbt2);

    binconv_wmma_kernel<<<gemmGrid, 256, 0, stream>>>(xb, wbt1, nullptr, outf);
    channel_stats_kernel<<<CCH, 256, 0, stream>>>(outf, stats);
    bn_binarize_kernel<<<padBlocks, 256, 0, stream>>>(outf, stats, g1, b1, xb);

    binconv_wmma_kernel<<<gemmGrid, 256, 0, stream>>>(xb, wbt2, x, outf);
    channel_stats_kernel<<<CCH, 256, 0, stream>>>(outf, stats);
    bn_out_kernel<<<(MTOT * CCH) / 256, 256, 0, stream>>>(outf, stats, g2, b2, out);
}
